// Model_30434138260182
// MI455X (gfx1250) — matrix operand order repaired and checked by reading, not compiled, not run on hardware
//
#include <hip/hip_runtime.h>

// ---------------------------------------------------------------------------
// MI455X (gfx1250, wave32) fused Mamba-forecaster.
// One workgroup (256 thr = 8 waves) per (batch, channel) sequence; all
// activations live in LDS; all GEMMs use v_wmma_f32_16x16x32_f16.
// Per-wave M-tile parity is constant -> A fragments hoisted to registers.
// ---------------------------------------------------------------------------

typedef _Float16 v16h __attribute__((ext_vector_type(16)));
typedef _Float16 v8h  __attribute__((ext_vector_type(8)));
typedef float    v8f  __attribute__((ext_vector_type(8)));

// f16 weight arena offsets (in halves) inside d_ws
#define WF_O_IN   0         // 4 x [512][128]
#define WF_O_XP   262144    // 4 x [48][256]   (N 40->48 zero pad)
#define WF_O_DT   311296    // 4 x [256][32]   (K 8->32 zero pad)
#define WF_O_OUT  344064    // 4 x [128][256]
#define WF_O_PE   475136    // [128][32]       (K 16->32 zero pad)
#define WF_O_D2P  479232    // [16][128]
#define WF_TOTAL  481280

// dynamic LDS byte offsets
#define SO_SEQ 0        // 512  f32 (normed series, later d2p output series)
#define SO_H   2048     // 32x128 f32 residual stream
#define SO_XZ  18432    // 32x512 f32 in_proj out (xi | z)
#define SO_XC  83968    // 32x256 f16 conv+silu out (= GEMM2 A staging)
#define SO_DT  100352   // 32x256 f16 softplus(dt)
#define SO_DBL 116736   // 32x48  f32 x_proj out (dt_raw | B | C)
#define SO_A16 122880   // 32x256 f16 generic A staging (norm out / scan y)
#define SO_DTR 139264   // 32x32  f16 dt_raw staging (K-padded)
#define SO_RED 141312   // 16 f32
#define SO_MS  141568   // 2  f32
#define SMEM_BYTES 141824

__device__ __forceinline__ v8f wmma_f16(v16h a, v16h b, v8f c) {
  return __builtin_amdgcn_wmma_f32_16x16x32_f16(false, a, false, b, (short)0, c,
                                                false, false);
}

// A fragment: 16x32 f16 tile from row-major [32][K] LDS buffer.
__device__ __forceinline__ v16h load_afrag(const _Float16* A, int K, int mt,
                                           int kt, int lane) {
  int row = mt * 16 + (lane & 15);
  int kb  = kt * 32 + ((lane >> 4) & 1) * 16;
  const _Float16* p = A + row * K + kb;
  union { v16h v; v8h h[2]; } u;
  u.h[0] = *(const v8h*)(p);
  u.h[1] = *(const v8h*)(p + 8);
  return u.v;
}

// B fragment: 32x16 f16 tile, B[k][n] = W[n][k], W row-major [N][K] (global f16).
__device__ __forceinline__ v16h load_bfrag(const _Float16* W, int K, int nt,
                                           int kt, int lane) {
  int n  = nt * 16 + (lane & 15);
  int kb = kt * 32 + ((lane >> 4) & 1) * 16;
  return *(const v16h*)(W + n * K + kb);
}

__device__ __forceinline__ void rmsnorm_to_a16(const float* sH, const float* w,
                                               _Float16* dst, int wave, int lane) {
  for (int r = wave; r < 32; r += 8) {
    float x0 = sH[r * 128 + lane];
    float x1 = sH[r * 128 + lane + 32];
    float x2 = sH[r * 128 + lane + 64];
    float x3 = sH[r * 128 + lane + 96];
    float ss = x0 * x0 + x1 * x1 + x2 * x2 + x3 * x3;
    for (int o = 16; o; o >>= 1) ss += __shfl_xor(ss, o, 32);
    float sc = rsqrtf(ss * (1.f / 128.f) + 1e-5f);
    dst[r * 128 + lane]      = (_Float16)(x0 * sc * w[lane]);
    dst[r * 128 + lane + 32] = (_Float16)(x1 * sc * w[lane + 32]);
    dst[r * 128 + lane + 64] = (_Float16)(x2 * sc * w[lane + 64]);
    dst[r * 128 + lane + 96] = (_Float16)(x3 * sc * w[lane + 96]);
  }
}

// ---------------------------------------------------------------------------
// Prep: fp32 weights -> f16 arena (with zero padding for WMMA tile shapes)
// ---------------------------------------------------------------------------
__global__ void prep_weights(const float* __restrict__ in_w,
                             const float* __restrict__ xp_w,
                             const float* __restrict__ dt_w,
                             const float* __restrict__ out_w,
                             const float* __restrict__ pe_w,
                             const float* __restrict__ d2p_w,
                             _Float16* __restrict__ ws) {
  int i = blockIdx.x * blockDim.x + threadIdx.x;
  if (i >= WF_TOTAL) return;
  float v;
  if (i < WF_O_XP) {
    v = in_w[i];
  } else if (i < WF_O_DT) {
    int r = i - WF_O_XP;
    int l = r / 12288, inner = r % 12288;
    int n = inner / 256, k = inner % 256;
    v = (n < 40) ? xp_w[(l * 40 + n) * 256 + k] : 0.f;
  } else if (i < WF_O_OUT) {
    int r = i - WF_O_DT;
    int l = r / 8192, inner = r % 8192;
    int n = inner / 32, k = inner % 32;
    v = (k < 8) ? dt_w[(l * 256 + n) * 8 + k] : 0.f;
  } else if (i < WF_O_PE) {
    v = out_w[i - WF_O_OUT];
  } else if (i < WF_O_D2P) {
    int r = i - WF_O_PE;
    int d = r / 32, k = r % 32;
    v = (k < 16) ? pe_w[d * 16 + k] : 0.f;
  } else {
    v = d2p_w[i - WF_O_D2P];
  }
  ws[i] = (_Float16)v;
}

// ---------------------------------------------------------------------------
// Main fused kernel: one block per (b, c)
// ---------------------------------------------------------------------------
__global__ __launch_bounds__(256)
void mamba_forecast(const float* __restrict__ x_enc,
                    const float* __restrict__ patch_emb_b,
                    const float* __restrict__ d2p_b,
                    const float* __restrict__ out_lin_w,
                    const float* __restrict__ out_lin_b,
                    const float* __restrict__ norm_w,
                    const float* __restrict__ block_norm_w,
                    const float* __restrict__ conv_w,
                    const float* __restrict__ conv_b,
                    const float* __restrict__ dt_proj_b,
                    const float* __restrict__ A_log,
                    const float* __restrict__ D_param,
                    const _Float16* __restrict__ wf,
                    float* __restrict__ out) {
  extern __shared__ __align__(128) char smem[];
  float*     sSeq = (float*)(smem + SO_SEQ);
  float*     sH   = (float*)(smem + SO_H);
  float*     sXZ  = (float*)(smem + SO_XZ);
  _Float16*  sXC  = (_Float16*)(smem + SO_XC);
  _Float16*  sDTp = (_Float16*)(smem + SO_DT);
  float*     sDBL = (float*)(smem + SO_DBL);
  _Float16*  sA16 = (_Float16*)(smem + SO_A16);
  _Float16*  sDTR = (_Float16*)(smem + SO_DTR);
  float*     sRed = (float*)(smem + SO_RED);
  float*     sMS  = (float*)(smem + SO_MS);

  const int tid  = threadIdx.x;
  // Scalarize the wave id: tile loops become SALU-controlled, EXEC stays all-1
  // around every v_wmma (required by the ISA) with no divergence bookkeeping.
  const int wave = __builtin_amdgcn_readfirstlane(tid >> 5);
  const int lane = tid & 31;
  const int mt   = wave & 1;        // per-wave constant M-tile parity
  const int ntb  = wave >> 1;       // base N-tile
  const int bc = blockIdx.x;
  const int b = bc >> 7, c = bc & 127;

  // ---- per-(b,c) mean/std over SEQ=512, normalize into sSeq -------------
  float x0 = x_enc[((size_t)b * 512 + tid) * 128 + c];
  float x1 = x_enc[((size_t)b * 512 + tid + 256) * 128 + c];
  float s = x0 + x1, q = x0 * x0 + x1 * x1;
  for (int o = 16; o; o >>= 1) { s += __shfl_xor(s, o, 32); q += __shfl_xor(q, o, 32); }
  if (lane == 0) { sRed[wave * 2] = s; sRed[wave * 2 + 1] = q; }
  __syncthreads();
  if (tid == 0) {
    float ts = 0.f, tq = 0.f;
    for (int w = 0; w < 8; ++w) { ts += sRed[2 * w]; tq += sRed[2 * w + 1]; }
    float mean = ts * (1.f / 512.f);
    float var  = tq * (1.f / 512.f) - mean * mean;
    sMS[0] = mean;
    sMS[1] = sqrtf(var + 1e-5f);
  }
  __syncthreads();
  const float mean = sMS[0], stdv = sMS[1], inv = 1.f / stdv;
  sSeq[tid]       = (x0 - mean) * inv;
  sSeq[tid + 256] = (x1 - mean) * inv;
  __syncthreads();

  // ---- stage patches as [32][32] f16 (K padded 16->32) -------------------
  #pragma unroll
  for (int i = 0; i < 4; ++i) {
    int slot = tid + i * 256;
    int t = slot >> 5, k = slot & 31;
    sA16[slot] = (k < 16) ? (_Float16)sSeq[t * 16 + k] : (_Float16)0.f;
  }
  __syncthreads();

  // ---- patch embedding: [32x32] x [32x128] -> sH -------------------------
  {
    const _Float16* W = wf + WF_O_PE;
    v16h a0 = load_afrag(sA16, 32, mt, 0, lane);
    #pragma unroll
    for (int i = 0; i < 2; ++i) {                 // 16 tiles / 8 waves
      int nt = ntb + 4 * i;
      v8f acc = {};
      acc = wmma_f16(a0, load_bfrag(W, 32, nt, 0, lane), acc);
      int col = nt * 16 + (lane & 15);
      float bias = patch_emb_b[col];
      int rb = mt * 16 + ((lane >> 4) & 1) * 8;
      #pragma unroll
      for (int v = 0; v < 8; ++v) sH[(rb + v) * 128 + col] = acc[v] + bias;
    }
  }
  __syncthreads();

  // ---- 4 Mamba layers ----------------------------------------------------
  for (int L = 0; L < 4; ++L) {
    rmsnorm_to_a16(sH, block_norm_w + L * 128, sA16, wave, lane);
    __syncthreads();

    // in_proj: [32x128] x [128x512] -> sXZ (xi | z)
    {
      const _Float16* W = wf + WF_O_IN + L * 512 * 128;
      v16h af[4];
      #pragma unroll
      for (int kt = 0; kt < 4; ++kt) af[kt] = load_afrag(sA16, 128, mt, kt, lane);
      #pragma unroll 2
      for (int i = 0; i < 8; ++i) {               // 64 tiles / 8 waves
        int nt = ntb + 4 * i;
        v8f acc = {};
        #pragma unroll
        for (int kt = 0; kt < 4; ++kt)
          acc = wmma_f16(af[kt], load_bfrag(W, 128, nt, kt, lane), acc);
        int col = nt * 16 + (lane & 15);
        int rb = mt * 16 + ((lane >> 4) & 1) * 8;
        #pragma unroll
        for (int v = 0; v < 8; ++v) sXZ[(rb + v) * 512 + col] = acc[v];
      }
    }
    __syncthreads();

    // causal depthwise conv (D_CONV=4) + silu ; thread = channel d
    {
      int d = tid;
      float w0 = conv_w[(L * 256 + d) * 4 + 0];
      float w1 = conv_w[(L * 256 + d) * 4 + 1];
      float w2 = conv_w[(L * 256 + d) * 4 + 2];
      float w3 = conv_w[(L * 256 + d) * 4 + 3];
      float cb = conv_b[L * 256 + d];
      float p0 = 0.f, p1 = 0.f, p2 = 0.f;
      for (int t = 0; t < 32; ++t) {
        float cur = sXZ[t * 512 + d];
        float v = w0 * p0 + w1 * p1 + w2 * p2 + w3 * cur + cb;
        float sv = v / (1.f + __expf(-v));          // silu
        sXC[t * 256 + d] = (_Float16)sv;
        p0 = p1; p1 = p2; p2 = cur;
      }
    }
    __syncthreads();

    // x_proj: [32x256] x [256x48] -> sDBL (dt_raw 0..7 | B 8..23 | C 24..39)
    if (wave < 6) {                               // 6 tiles, 1 per wave
      const _Float16* W = wf + WF_O_XP + L * 48 * 256;
      v8f acc = {};
      #pragma unroll
      for (int kt = 0; kt < 8; ++kt) {
        v16h a = load_afrag(sXC, 256, mt, kt, lane);
        acc = wmma_f16(a, load_bfrag(W, 256, ntb, kt, lane), acc);
      }
      int col = ntb * 16 + (lane & 15);
      int rb = mt * 16 + ((lane >> 4) & 1) * 8;
      #pragma unroll
      for (int v = 0; v < 8; ++v) sDBL[(rb + v) * 48 + col] = acc[v];
    }
    __syncthreads();

    // stage dt_raw [32][32] f16 (K padded 8->32)
    #pragma unroll
    for (int i = 0; i < 4; ++i) {
      int slot = tid + i * 256;
      int t = slot >> 5, k = slot & 31;
      sDTR[slot] = (k < 8) ? (_Float16)sDBL[t * 48 + k] : (_Float16)0.f;
    }
    __syncthreads();

    // dt_proj: [32x32] x [32x256] + bias, softplus -> sDTp
    {
      const _Float16* W = wf + WF_O_DT + L * 256 * 32;
      v16h a0 = load_afrag(sDTR, 32, mt, 0, lane);
      #pragma unroll 2
      for (int i = 0; i < 4; ++i) {               // 32 tiles / 8 waves
        int nt = ntb + 4 * i;
        v8f acc = {};
        acc = wmma_f16(a0, load_bfrag(W, 32, nt, 0, lane), acc);
        int col = nt * 16 + (lane & 15);
        float bias = dt_proj_b[L * 256 + col];
        int rb = mt * 16 + ((lane >> 4) & 1) * 8;
        #pragma unroll
        for (int v = 0; v < 8; ++v) {
          float xx = acc[v] + bias;
          float sp = (xx > 20.f) ? xx : log1pf(__expf(xx));
          sDTp[(rb + v) * 256 + col] = (_Float16)sp;
        }
      }
    }
    __syncthreads();

    // selective scan: thread = channel d, 16-state in VGPRs
    {
      int d = tid;
      float a[16], hs[16];
      #pragma unroll
      for (int si = 0; si < 16; ++si) {
        a[si]  = -__expf(A_log[(L * 256 + d) * 16 + si]);
        hs[si] = 0.f;
      }
      float Dp = D_param[L * 256 + d];
      for (int t = 0; t < 32; ++t) {
        float dtv = (float)sDTp[t * 256 + d];
        float xv  = (float)sXC[t * 256 + d];
        float zv  = sXZ[t * 512 + 256 + d];
        const float4* Bp = (const float4*)(sDBL + t * 48 + 8);
        const float4* Cp = (const float4*)(sDBL + t * 48 + 24);
        float y = 0.f;
        float dx = dtv * xv;
        #pragma unroll
        for (int g = 0; g < 4; ++g) {
          union { float4 v; float f[4]; } Bv, Cv;
          Bv.v = Bp[g]; Cv.v = Cp[g];
          #pragma unroll
          for (int j = 0; j < 4; ++j) {
            int si = g * 4 + j;
            float dA = __expf(dtv * a[si]);
            hs[si] = dA * hs[si] + dx * Bv.f[j];
            y += hs[si] * Cv.f[j];
          }
        }
        float yf = (y + xv * Dp) * (zv / (1.f + __expf(-zv)));
        sA16[t * 256 + d] = (_Float16)yf;   // GEMM4 A staging
      }
    }
    __syncthreads();

    // out_proj: [32x256] x [256x128] -> residual add into sH
    {
      const _Float16* W = wf + WF_O_OUT + L * 128 * 256;
      v16h af[8];
      #pragma unroll
      for (int kt = 0; kt < 8; ++kt) af[kt] = load_afrag(sA16, 256, mt, kt, lane);
      #pragma unroll
      for (int i = 0; i < 2; ++i) {               // 16 tiles / 8 waves
        int nt = ntb + 4 * i;
        v8f acc = {};
        #pragma unroll
        for (int kt = 0; kt < 8; ++kt)
          acc = wmma_f16(af[kt], load_bfrag(W, 256, nt, kt, lane), acc);
        int col = nt * 16 + (lane & 15);
        int rb = mt * 16 + ((lane >> 4) & 1) * 8;
        #pragma unroll
        for (int v = 0; v < 8; ++v) sH[(rb + v) * 128 + col] += acc[v];
      }
    }
    __syncthreads();
  }

  // ---- final rmsnorm -> sA16 [32][128] -----------------------------------
  rmsnorm_to_a16(sH, norm_w, sA16, wave, lane);
  __syncthreads();

  // ---- d2p head: [32x128] x [128x16] -> series sSeq[512] -----------------
  if (wave < 2) {                                 // 2 tiles (mt = wave), nt = 0
    const _Float16* W = wf + WF_O_D2P;
    v8f acc = {};
    #pragma unroll
    for (int kt = 0; kt < 4; ++kt) {
      v16h a = load_afrag(sA16, 128, wave, kt, lane);
      acc = wmma_f16(a, load_bfrag(W, 128, 0, kt, lane), acc);
    }
    int col = (lane & 15);
    float bias = d2p_b[col];
    int rb = wave * 16 + ((lane >> 4) & 1) * 8;
    #pragma unroll
    for (int v = 0; v < 8; ++v) sSeq[(rb + v) * 16 + col] = acc[v] + bias;
  }
  __syncthreads();

  // ---- out_lin (96x512 GEMV) + de-norm + strided store -------------------
  if (wave < 6) {                                 // 192 threads, 2 per output
    int p = tid >> 1, h = tid & 1;
    const float4* wr = (const float4*)(out_lin_w + p * 512 + h * 256);
    const float4* sv = (const float4*)(sSeq + h * 256);
    float acc = 0.f;
    for (int i = 0; i < 64; ++i) {
      float4 wv = wr[i], xv = sv[i];
      acc += wv.x * xv.x + wv.y * xv.y + wv.z * xv.z + wv.w * xv.w;
    }
    acc += __shfl_xor(acc, 1, 32);                // combine the two halves
    if (h == 0) {
      float val = (acc + out_lin_b[p]) * stdv + mean;
      out[((size_t)b * 96 + p) * 128 + c] = val;
    }
  }
}

// ---------------------------------------------------------------------------
extern "C" void kernel_launch(void* const* d_in, const int* in_sizes, int n_in,
                              void* d_out, int out_size, void* d_ws, size_t ws_size,
                              hipStream_t stream) {
  const float* x_enc        = (const float*)d_in[0];
  // d_in[1] = x_mark_enc (unused by reference path)
  const float* patch_emb_w  = (const float*)d_in[2];
  const float* patch_emb_b  = (const float*)d_in[3];
  const float* d2p_w        = (const float*)d_in[4];
  const float* d2p_b        = (const float*)d_in[5];
  const float* out_lin_w    = (const float*)d_in[6];
  const float* out_lin_b    = (const float*)d_in[7];
  const float* norm_w       = (const float*)d_in[8];
  const float* block_norm_w = (const float*)d_in[9];
  const float* in_proj_w    = (const float*)d_in[10];
  const float* conv_w       = (const float*)d_in[11];
  const float* conv_b       = (const float*)d_in[12];
  const float* x_proj_w     = (const float*)d_in[13];
  const float* dt_proj_w    = (const float*)d_in[14];
  const float* dt_proj_b    = (const float*)d_in[15];
  const float* A_log        = (const float*)d_in[16];
  const float* D_param      = (const float*)d_in[17];
  const float* out_proj_w   = (const float*)d_in[18];
  _Float16* wf16 = (_Float16*)d_ws;
  float* out = (float*)d_out;

  (void)hipFuncSetAttribute((const void*)mamba_forecast,
                            hipFuncAttributeMaxDynamicSharedMemorySize,
                            SMEM_BYTES);

  prep_weights<<<(WF_TOTAL + 255) / 256, 256, 0, stream>>>(
      in_proj_w, x_proj_w, dt_proj_w, out_proj_w, patch_emb_w, d2p_w, wf16);

  mamba_forecast<<<4096, 256, SMEM_BYTES, stream>>>(
      x_enc, patch_emb_b, d2p_b, out_lin_w, out_lin_b, norm_w, block_norm_w,
      conv_w, conv_b, dt_proj_b, A_log, D_param, wf16, out);
}
